// FEBf_32057635897484
// MI455X (gfx1250) — matrix operand order repaired and checked by reading, not compiled, not run on hardware
//
#include <hip/hip_runtime.h>

// Problem constants
#define BATCH   16
#define SEQL    4096
#define NHEAD   16
#define DHEAD   128
#define NMODE   64
#define NC      128   // 2*NMODE (re || im)

typedef __attribute__((ext_vector_type(16))) __bf16 v16bf;
typedef __attribute__((ext_vector_type(8)))  __bf16 v8bf;
typedef __attribute__((ext_vector_type(4)))  __bf16 v4bf;
typedef __attribute__((ext_vector_type(8)))  float  v8f;

// ---------------------------------------------------------------------------
// Fragment assembly helpers (CDNA5 16-bit WMMA VGPR layouts, wave32)
// A (16x32, row-major source): lane L<16 -> row M=L, halves [K0..7 | K16..23]
//                              lane L>=16 -> row M=L-16, halves [K8..15 | K24..31]
// B (32x16, column-major source): lane L<16 -> col N=L, halves [K0..7 | K16..23]
//                              lane L>=16 -> col N=L-16, halves [K8..15 | K24..31]
// ---------------------------------------------------------------------------
__device__ __forceinline__ v16bf load_a_frag(const __bf16* rowptr, int k0, int hi) {
  const __bf16* p = rowptr + k0 + (hi ? 8 : 0);
  union { v16bf v; v8bf h[2]; } u;
  u.h[0] = *(const v8bf*)(p);
  u.h[1] = *(const v8bf*)(p + 16);
  return u.v;
}
__device__ __forceinline__ v16bf load_b_frag(const __bf16* colptr, int k0, int hi) {
  const __bf16* p = colptr + k0 + (hi ? 8 : 0);
  union { v16bf v; v8bf h[2]; } u;
  u.h[0] = *(const v8bf*)(p);
  u.h[1] = *(const v8bf*)(p + 16);
  return u.v;
}
#define WMMA_BF16(A, B, C) \
  __builtin_amdgcn_wmma_f32_16x16x32_bf16(false, (A), false, (B), (short)0, (C), false, false)

// ---------------------------------------------------------------------------
// Kernel 0: precompute truncated Fourier bases in bf16.
//   Af[c][t]  (forward): c<64 -> cos(2*pi*c*t/T), c>=64 -> -sin(...)
//   Gt[tau][c] (inverse): scaled by 1/T (DC) or 2/T (irfft with modes >= 64
//     zeroed; Im of bin 0 discarded by the C2R transform).
// Angle reduced exactly via (m*t) mod 4096 -> full float precision.
// ---------------------------------------------------------------------------
__global__ void basis_kernel(__bf16* __restrict__ Af, __bf16* __restrict__ Gt) {
  int idx = blockIdx.x * 256 + threadIdx.x;     // 128 * 4096 entries
  int t = idx & (SEQL - 1);
  int c = idx >> 12;
  int m = c & (NMODE - 1);
  bool isim = c >= NMODE;
  int phase = (m * t) & (SEQL - 1);
  float theta = (float)phase * (6.283185307179586f / (float)SEQL);
  float sn = __sinf(theta);
  float cs = __cosf(theta);
  Af[(size_t)c * SEQL + t] = (__bf16)(isim ? -sn : cs);
  float g;
  if (isim) g = (-2.0f / (float)SEQL) * sn;
  else      g = ((m == 0 ? 1.0f : 2.0f) / (float)SEQL) * cs;
  Gt[(size_t)t * NC + c] = (__bf16)g;
}

// ---------------------------------------------------------------------------
// Kernel 1: weight transpose  w[h][i][o][m] (f32) -> Wt[h][m][o][i] (bf16)
// LDS-tiled (64 i x 64 m per block): global read coalesced in m, global
// write coalesced in i.
// ---------------------------------------------------------------------------
__global__ __launch_bounds__(256) void wtrans_kernel(const float* __restrict__ Win,
                                                     __bf16* __restrict__ Wt) {
  int ho = blockIdx.x;            // h*128 + o
  int h = ho >> 7, o = ho & 127;
  int i0 = blockIdx.y * 64;
  __shared__ __align__(16) __bf16 ls[64][66];
  int tid = threadIdx.x;
  int mcol = tid & 63, rr = tid >> 6;
#pragma unroll
  for (int r = 0; r < 16; ++r) {
    int il = r * 4 + rr;
    float v = Win[(((size_t)h * DHEAD + (i0 + il)) * DHEAD + o) * NMODE + mcol];
    ls[il][mcol] = (__bf16)v;
  }
  __syncthreads();
#pragma unroll
  for (int r = 0; r < 16; ++r) {
    int ml = r * 4 + rr;
    Wt[(((size_t)h * NMODE + ml) * DHEAD + o) * DHEAD + (i0 + mcol)] = ls[mcol][ml];
  }
}

// ---------------------------------------------------------------------------
// Kernel 2: forward projection GEMM (per b,h):  S[c][i] = sum_t Af[c][t]*x[t][i]
// M=128 (c), N=128 (i), K=4096. Software-pipelined with a double-buffered
// LDS x tile: global loads for tile k+1 are issued before tile k's WMMAs,
// conversion + ds_store land in the alternate buffer, one barrier/iter.
// All 8 B fragments are preloaded into distinct registers so the DS waits
// retire progressively instead of draining before each WMMA.
// Output S[b][h][c][i] bf16 (A-operand layout for the mixing GEMM).
// ---------------------------------------------------------------------------
__global__ __launch_bounds__(256) void gemm1_kernel(const float* __restrict__ x,
                                                    const __bf16* __restrict__ Af,
                                                    __bf16* __restrict__ S) {
  int b = blockIdx.x, h = blockIdx.y;
  __shared__ __align__(16) __bf16 xs[2][128][32];   // [buf][i][t], 16 KB
  int tid = threadIdx.x;
  int wave = tid >> 5, lane = tid & 31;
  int hi = lane >> 4, ln = lane & 15;
  int crow = wave * 16 + ln;                        // A row this lane supplies

  v8f acc[8];
#pragma unroll
  for (int n = 0; n < 8; ++n) acc[n] = v8f{};

  const float* xb = x + (size_t)b * SEQL * NHEAD * DHEAD + (size_t)h * DHEAD;
  const __bf16* Arow = Af + (size_t)crow * SEQL;

  int ii = tid & 127;               // staging: column this thread converts
  int tg = (tid >> 7) * 4;          // 0 or 4
  const float* xcol = xb + ii;

  // Prologue: stage tile 0 into buffer 0.
#pragma unroll
  for (int e = 0; e < 4; ++e) {
    int t0 = e * 8 + tg;
    v4bf pk = {(__bf16)xcol[(size_t)(t0 + 0) * (NHEAD * DHEAD)],
               (__bf16)xcol[(size_t)(t0 + 1) * (NHEAD * DHEAD)],
               (__bf16)xcol[(size_t)(t0 + 2) * (NHEAD * DHEAD)],
               (__bf16)xcol[(size_t)(t0 + 3) * (NHEAD * DHEAD)]};
    *(v4bf*)(&xs[0][ii][t0]) = pk;
  }
  __syncthreads();

  for (int kt = 0; kt < SEQL; kt += 32) {
    int cur = (kt >> 5) & 1;
    bool more = (kt + 32) < SEQL;

    // Issue next tile's global loads early (no waits until conversion).
    float f[16];
    if (more) {
      const float* xk = xcol + (size_t)(kt + 32) * (NHEAD * DHEAD);
#pragma unroll
      for (int e = 0; e < 16; ++e)
        f[e] = xk[(size_t)((e >> 2) * 8 + tg + (e & 3)) * (NHEAD * DHEAD)];
    }

    // Compute on current buffer: preload all B fragments, then 8 WMMAs.
    v16bf a = load_a_frag(Arow, kt, hi);
    v16bf bf[8];
#pragma unroll
    for (int n = 0; n < 8; ++n)
      bf[n] = load_b_frag(&xs[cur][n * 16 + ln][0], 0, hi);
#pragma unroll
    for (int n = 0; n < 8; ++n)
      acc[n] = WMMA_BF16(a, bf[n], acc[n]);

    // Convert + store next tile into the alternate buffer.
    if (more) {
#pragma unroll
      for (int e = 0; e < 4; ++e) {
        int t0 = e * 8 + tg;
        v4bf pk = {(__bf16)f[e * 4 + 0], (__bf16)f[e * 4 + 1],
                   (__bf16)f[e * 4 + 2], (__bf16)f[e * 4 + 3]};
        *(v4bf*)(&xs[cur ^ 1][ii][t0]) = pk;
      }
      if (kt + 64 < SEQL) {         // near-cache prefetch two tiles ahead
        int pr = tid >> 3;
        int pc = (tid & 7) * 16;
        __builtin_prefetch(xb + (size_t)(kt + 64 + pr) * (NHEAD * DHEAD) + pc, 0, 3);
      }
    }
    __syncthreads();
  }

  __bf16* Sp = S + ((size_t)b * NHEAD + h) * NC * DHEAD;
#pragma unroll
  for (int n = 0; n < 8; ++n) {
    int icol = n * 16 + ln;
#pragma unroll
    for (int r = 0; r < 8; ++r) {
      int c = wave * 16 + r + hi * 8;
      Sp[(size_t)c * DHEAD + icol] = (__bf16)acc[n][r];
    }
  }
}

// ---------------------------------------------------------------------------
// Kernel 3: complex mode mixing. Block = (h, m): complex GEMM M=16 (batch),
// K=128 (i), N=128 (o) as 4 real bf16 WMMA chains:
//   re = A_re*B_re - A_im*B_im   (two accumulators, one v8f subtract;
//                                 avoids the bf16-illegal A/B NEG bits)
//   im = A_re*B_im + A_im*B_re   (two WMMAs into one accumulator)
// B fragments come straight from the pre-transposed bf16 weights (each block
// reads a unique 64 KB tile once -> no LDS reuse to exploit).
// Output Mo[b][h][o][c] bf16 (B-operand layout for GEMM2).
// ---------------------------------------------------------------------------
__global__ __launch_bounds__(256) void mix_kernel(const __bf16* __restrict__ S,
                                                  const __bf16* __restrict__ Wre,
                                                  const __bf16* __restrict__ Wim,
                                                  __bf16* __restrict__ Mo) {
  int h = blockIdx.x, m = blockIdx.y;
  int tid = threadIdx.x;
  int wave = tid >> 5, lane = tid & 31;
  int hi = lane >> 4, ln = lane & 15;
  int otile = wave * 16;

  v8f arr = v8f{}, aii = v8f{}, aim = v8f{};

  const __bf16* Sre = S + (((size_t)ln * NHEAD + h) * NC + m) * DHEAD;
  const __bf16* Sim = S + (((size_t)ln * NHEAD + h) * NC + NMODE + m) * DHEAD;
  const __bf16* Bre = Wre + (((size_t)h * NMODE + m) * DHEAD + otile + ln) * DHEAD;
  const __bf16* Bim = Wim + (((size_t)h * NMODE + m) * DHEAD + otile + ln) * DHEAD;

#pragma unroll
  for (int k = 0; k < DHEAD; k += 32) {
    v16bf are = load_a_frag(Sre, k, hi);
    v16bf aimf = load_a_frag(Sim, k, hi);
    v16bf bre = load_b_frag(Bre, k, hi);
    v16bf bim = load_b_frag(Bim, k, hi);
    arr = WMMA_BF16(are, bre, arr);
    aii = WMMA_BF16(aimf, bim, aii);
    aim = WMMA_BF16(are, bim, aim);
    aim = WMMA_BF16(aimf, bre, aim);
  }
  v8f dre = arr - aii;

#pragma unroll
  for (int r = 0; r < 8; ++r) {
    int bb = r + hi * 8;                                  // batch row
    size_t base = (((size_t)bb * NHEAD + h) * DHEAD + (otile + ln)) * NC;
    Mo[base + m]         = (__bf16)dre[r];
    Mo[base + NMODE + m] = (__bf16)aim[r];
  }
}

// ---------------------------------------------------------------------------
// Kernel 4: inverse projection GEMM (per b,h,tau-block):
//   out[tau][o] = sum_c Gt[tau][c] * Mo[o][c]
// The 32 KB Mo[b][h] tile is staged into LDS with gfx1250 async LDS copies
// (global_load_async_to_lds_b128, tracked by ASYNCcnt), then all 8 waves
// build B fragments from LDS (preloaded per K-step so waits pipeline).
// Stores f32 directly to out[b][tau][h][o] (64 B coalesced per row segment).
// ---------------------------------------------------------------------------
__global__ __launch_bounds__(256) void gemm2_kernel(const __bf16* __restrict__ Gt,
                                                    const __bf16* __restrict__ Mo,
                                                    float* __restrict__ out) {
  int b = blockIdx.x, h = blockIdx.y, tb = blockIdx.z;
  int tid = threadIdx.x;
  int wave = tid >> 5, lane = tid & 31;
  int hi = lane >> 4, ln = lane & 15;
  int trow = tb * 128 + wave * 16 + ln;

  __shared__ __align__(16) __bf16 ms[DHEAD * NC];   // Mo tile [o][c], 32 KB

  const __bf16* Arow = Gt + (size_t)trow * NC;
  const __bf16* Mbase = Mo + ((size_t)b * NHEAD + h) * DHEAD * NC;

  // Async-copy the 32 KB tile: 256 threads x 8 x 16B = 32 KB.
  unsigned lds0 = (unsigned)(uintptr_t)(&ms[0]);    // LDS offset = addr[31:0]
#pragma unroll
  for (int e = 0; e < 8; ++e) {
    unsigned off = (unsigned)(e * 4096 + tid * 16);
    unsigned ldsa = lds0 + off;
    const void* ga = (const char*)Mbase + off;
    asm volatile("global_load_async_to_lds_b128 %0, %1, off"
                 :: "v"(ldsa), "v"(ga) : "memory");
  }
  asm volatile("s_wait_asynccnt 0x0" ::: "memory");
  __syncthreads();

  v8f acc[8];
#pragma unroll
  for (int n = 0; n < 8; ++n) acc[n] = v8f{};

#pragma unroll
  for (int k = 0; k < NC; k += 32) {
    v16bf a = load_a_frag(Arow, k, hi);
    v16bf bf[8];
#pragma unroll
    for (int n = 0; n < 8; ++n)
      bf[n] = load_b_frag(&ms[(size_t)(n * 16 + ln) * NC], k, hi);
#pragma unroll
    for (int n = 0; n < 8; ++n)
      acc[n] = WMMA_BF16(a, bf[n], acc[n]);
  }

#pragma unroll
  for (int n = 0; n < 8; ++n) {
    int o = n * 16 + ln;
#pragma unroll
    for (int r = 0; r < 8; ++r) {
      int tau = tb * 128 + wave * 16 + r + hi * 8;
      out[(((size_t)b * SEQL + tau) * NHEAD + h) * DHEAD + o] = acc[n][r];
    }
  }
}

// ---------------------------------------------------------------------------
extern "C" void kernel_launch(void* const* d_in, const int* in_sizes, int n_in,
                              void* d_out, int out_size, void* d_ws, size_t ws_size,
                              hipStream_t stream) {
  (void)in_sizes; (void)n_in; (void)out_size; (void)ws_size;
  const float* x   = (const float*)d_in[0];
  const float* wre = (const float*)d_in[1];
  const float* wim = (const float*)d_in[2];
  float* out = (float*)d_out;

  // Workspace partition (all bf16):
  char* ws = (char*)d_ws;
  __bf16* Af   = (__bf16*)ws; ws += (size_t)NC * SEQL * 2;                      // 1 MB
  __bf16* Gt   = (__bf16*)ws; ws += (size_t)SEQL * NC * 2;                      // 1 MB
  __bf16* Wtre = (__bf16*)ws; ws += (size_t)NHEAD * NMODE * DHEAD * DHEAD * 2;  // 33.5 MB
  __bf16* Wtim = (__bf16*)ws; ws += (size_t)NHEAD * NMODE * DHEAD * DHEAD * 2;  // 33.5 MB
  __bf16* S    = (__bf16*)ws; ws += (size_t)BATCH * NHEAD * NC * DHEAD * 2;     // 8.4 MB
  __bf16* Mo   = (__bf16*)ws; ws += (size_t)BATCH * NHEAD * DHEAD * NC * 2;     // 8.4 MB

  basis_kernel<<<dim3((NC * SEQL) / 256), dim3(256), 0, stream>>>(Af, Gt);
  wtrans_kernel<<<dim3(NHEAD * DHEAD, 2), dim3(256), 0, stream>>>(wre, Wtre);
  wtrans_kernel<<<dim3(NHEAD * DHEAD, 2), dim3(256), 0, stream>>>(wim, Wtim);
  gemm1_kernel<<<dim3(BATCH, NHEAD), dim3(256), 0, stream>>>(x, Af, S);
  mix_kernel<<<dim3(NHEAD, NMODE), dim3(256), 0, stream>>>(S, Wtre, Wtim, Mo);
  gemm2_kernel<<<dim3(BATCH, NHEAD, SEQL / 128), dim3(256), 0, stream>>>(Gt, Mo, out);
}
